// HierarchicalS2WithCopy_79517024518163
// MI455X (gfx1250) — matrix operand order repaired and checked by reading, not compiled, not run on hardware
//
#include <hip/hip_runtime.h>
#include <hip/hip_bf16.h>
#include <math.h>

// ---------------- problem constants ----------------
constexpr int B   = 256;
constexpr int T   = 100;
constexpr int D   = 256;
constexpr int H   = 8;
constexpr int DH  = 32;          // D / H
constexpr int HID = 128;
constexpr int V11 = 1000, V13 = 5000, V14 = 20000, VX = 40000;
constexpr int BT  = B * T;       // 25600

// ---------------- WMMA types (CDNA5, wave32) ----------------
typedef __attribute__((ext_vector_type(16))) __bf16 v16bf;
typedef __attribute__((ext_vector_type(8)))  float  v8f;

// =====================================================================
// bf16 GEMM: C[M,N] = act( A[M,K] @ W[N,K]^T + bias [+ res] )
//   A, W bf16 row-major (K contiguous): pure b128-load -> v_wmma loop.
//   One wave computes a 32x32 C macro-tile (2x2 WMMA tiles).
//   K-loop unrolled 2x (64/iter): 16 b128 loads feed 8 WMMAs per iter.
//   4 waves / block in a 2x2 grid cover 64(M) x 64(N).
//   Weight rows prefetched 2 iterations ahead (global_prefetch_b8).
//   Requires: M % 64 == 0, K % 64 == 0. N edge: clamp loads, guard stores.
// =====================================================================
template <bool HAS_RES, bool GELU>
__global__ __launch_bounds__(128)
void gemm_bf16_wmma(const __bf16* __restrict__ A, const __bf16* __restrict__ W,
                    const float* __restrict__ bias, const float* __restrict__ res,
                    float* __restrict__ C, int M, int N, int K) {
  const int lane = threadIdx.x & 31;
  const int wave = threadIdx.x >> 5;
  const int r  = lane & 15;
  const int hi = lane >> 4;

  const int mt0 = blockIdx.y * 64 + (wave >> 1) * 32;   // wave's 2 m-tiles
  const int nt0 = blockIdx.x * 64 + (wave & 1) * 32;    // wave's 2 n-tiles

  const int kaoff = hi * 8;    // A lanes 0-15 -> K {0..7,16..23}; 16-31 -> {8..15,24..31}
  const int kboff = hi * 8;    // B lanes 0-15 -> K {0..7,16..23}; 16-31 -> {8..15,24..31}

  const __bf16* __restrict__ Ar0 = A + (size_t)(mt0 + r) * K;
  const __bf16* __restrict__ Ar1 = A + (size_t)(mt0 + 16 + r) * K;
  const int gn0 = nt0 + r, gn1 = nt0 + 16 + r;
  const __bf16* __restrict__ Br0 = W + (size_t)(gn0 < N ? gn0 : N - 1) * K;
  const __bf16* __restrict__ Br1 = W + (size_t)(gn1 < N ? gn1 : N - 1) * K;

  v8f acc00 = {}, acc01 = {}, acc10 = {}, acc11 = {};

  for (int k0 = 0; k0 < K; k0 += 64) {
    // prefetch streaming weight rows two iterations ahead
    __builtin_prefetch(Br0 + k0 + kboff + 128, 0, 0);
    __builtin_prefetch(Br1 + k0 + kboff + 128, 0, 0);
#pragma unroll
    for (int kk = 0; kk < 64; kk += 32) {
      const int kb = k0 + kk;
      v16bf a0, a1, b0, b1;
#pragma unroll
      for (int j = 0; j < 8; ++j) {
        a0[j]     = Ar0[kb + kaoff + j];
        a0[8 + j] = Ar0[kb + kaoff + 16 + j];
        a1[j]     = Ar1[kb + kaoff + j];
        a1[8 + j] = Ar1[kb + kaoff + 16 + j];
      }
#pragma unroll
      for (int j = 0; j < 8; ++j) {
        b0[j]     = Br0[kb + kboff + j];
        b0[8 + j] = Br0[kb + kboff + 16 + j];
        b1[j]     = Br1[kb + kboff + j];
        b1[8 + j] = Br1[kb + kboff + 16 + j];
      }
      acc00 = __builtin_amdgcn_wmma_f32_16x16x32_bf16(false, a0, false, b0, (short)0, acc00, false, false);
      acc01 = __builtin_amdgcn_wmma_f32_16x16x32_bf16(false, a0, false, b1, (short)0, acc01, false, false);
      acc10 = __builtin_amdgcn_wmma_f32_16x16x32_bf16(false, a1, false, b0, (short)0, acc10, false, false);
      acc11 = __builtin_amdgcn_wmma_f32_16x16x32_bf16(false, a1, false, b1, (short)0, acc11, false, false);
    }
  }

  auto store_tile = [&](const v8f& acc, int mt, int nt) {
    const int n = nt + r;
    if (n < N) {
      const float bv = bias[n];
#pragma unroll
      for (int v = 0; v < 8; ++v) {
        const int m = mt + hi * 8 + v;
        float x = acc[v] + bv;
        if constexpr (HAS_RES) x += res[(size_t)m * N + n];
        if constexpr (GELU)    x = 0.5f * x * (1.0f + erff(x * 0.70710678118654752f));
        C[(size_t)m * N + n] = x;
      }
    }
  };
  store_tile(acc00, mt0,      nt0);
  store_tile(acc01, mt0,      nt0 + 16);
  store_tile(acc10, mt0 + 16, nt0);
  store_tile(acc11, mt0 + 16, nt0 + 16);
}

// f32 -> bf16 streaming convert (grid-stride)
__global__ void cvt_bf16_kernel(const float* __restrict__ in, __bf16* __restrict__ out,
                                size_t n) {
  for (size_t i = (size_t)blockIdx.x * blockDim.x + threadIdx.x; i < n;
       i += (size_t)gridDim.x * blockDim.x)
    out[i] = (__bf16)in[i];
}

// =====================================================================
// Multi-head attention (T=100, dh=32), softmax over keys with pad bias.
// =====================================================================
__global__ __launch_bounds__(128)
void attn_kernel(const float* __restrict__ qkv, const unsigned char* __restrict__ pad,
                 float* __restrict__ out) {
  const int bh = blockIdx.x;
  const int b = bh / H, h = bh % H;
  __shared__ float Ks[T][DH + 1];
  __shared__ float Vs[T][DH + 1];

  for (int i = threadIdx.x; i < T * DH; i += 128) {
    const int t = i >> 5, c = i & 31;
    const float* base = qkv + ((size_t)(b * T + t)) * (3 * D) + h * DH + c;
    Ks[t][c] = base[D];
    Vs[t][c] = base[2 * D];
  }
  __syncthreads();

  const int t = threadIdx.x;
  if (t < T) {
    float q[DH];
#pragma unroll
    for (int c = 0; c < DH; ++c)
      q[c] = qkv[((size_t)(b * T + t)) * (3 * D) + h * DH + c];
    const float scale = 0.17677669529663687f;   // 1/sqrt(32)

    float mx = -1e30f;
    for (int j = 0; j < T; ++j) {
      float d = 0.0f;
#pragma unroll
      for (int c = 0; c < DH; ++c) d += q[c] * Ks[j][c];
      d = d * scale + (pad[b * T + j] ? -1e9f : 0.0f);
      mx = fmaxf(mx, d);
    }
    float sum = 0.0f, accv[DH];
#pragma unroll
    for (int c = 0; c < DH; ++c) accv[c] = 0.0f;
    for (int j = 0; j < T; ++j) {
      float d = 0.0f;
#pragma unroll
      for (int c = 0; c < DH; ++c) d += q[c] * Ks[j][c];
      d = d * scale + (pad[b * T + j] ? -1e9f : 0.0f);
      const float w = expf(d - mx);
      sum += w;
#pragma unroll
      for (int c = 0; c < DH; ++c) accv[c] += w * Vs[j][c];
    }
    const float inv = 1.0f / sum;
#pragma unroll
    for (int c = 0; c < DH; ++c)
      out[((size_t)(b * T + t)) * D + h * DH + c] = accv[c] * inv;
  }
}

// =====================================================================
// LayerNorm over last dim (256). One wave per row, 8 rows per block.
// =====================================================================
__global__ __launch_bounds__(256)
void ln_kernel(const float* __restrict__ in, const float* __restrict__ g,
               const float* __restrict__ be, float* __restrict__ out) {
  const int lane = threadIdx.x & 31;
  const int wave = threadIdx.x >> 5;
  const size_t row = (size_t)blockIdx.x * 8 + wave;
  const float* r = in + row * D;
  float v[8];
  float s = 0.0f;
#pragma unroll
  for (int i = 0; i < 8; ++i) { v[i] = r[lane + 32 * i]; s += v[i]; }
#pragma unroll
  for (int m = 16; m; m >>= 1) s += __shfl_xor(s, m, 32);
  const float mean = s * (1.0f / D);
  float vs = 0.0f;
#pragma unroll
  for (int i = 0; i < 8; ++i) { const float d = v[i] - mean; vs += d * d; }
#pragma unroll
  for (int m = 16; m; m >>= 1) vs += __shfl_xor(vs, m, 32);
  const float inv = rsqrtf(vs * (1.0f / D) + 1e-5f);
#pragma unroll
  for (int i = 0; i < 8; ++i) {
    const int c = lane + 32 * i;
    out[row * D + c] = (v[i] - mean) * inv * g[c] + be[c];
  }
}

// ---------------- element-wise / builder kernels ----------------
__global__ void pe_kernel(float* __restrict__ pe) {
  const int t = blockIdx.x, c = threadIdx.x;
  const int i = c >> 1;
  const float div = expf((float)(2 * i) * (-logf(10000.0f) / (float)D));
  const float a = (float)t * div;
  pe[t * D + c] = (c & 1) ? cosf(a) : sinf(a);
}

__global__ void build_feat(const int* __restrict__ i11, const int* __restrict__ i13,
                           const int* __restrict__ i14, const int* __restrict__ ix,
                           const float* __restrict__ e11, const float* __restrict__ e13,
                           const float* __restrict__ e14, const float* __restrict__ ex,
                           float* __restrict__ feat) {
  const int bt = blockIdx.x, c = threadIdx.x;          // 384 threads
  float v;
  if (c < 64)        v = e11[(size_t)i11[bt] * 64 + c];
  else if (c < 128)  v = e13[(size_t)i13[bt] * 64 + (c - 64)];
  else if (c < 256)  v = e14[(size_t)i14[bt] * 128 + (c - 128)];
  else               v = ex [(size_t)ix [bt] * 128 + (c - 256)];
  feat[(size_t)bt * 384 + c] = v;
}

__global__ void add_pe_user(float* __restrict__ x, const float* __restrict__ pe,
                            const float* __restrict__ uemb, const int* __restrict__ useq) {
  const int bt = blockIdx.x, c = threadIdx.x;
  const int b = bt / T, t = bt % T;
  x[(size_t)bt * D + c] += pe[t * D + c] + uemb[(size_t)useq[b * T] * D + c];
}

__global__ void add_pe(float* __restrict__ x, const float* __restrict__ pe) {
  const int bt = blockIdx.x, c = threadIdx.x;
  x[(size_t)bt * D + c] += pe[(bt % T) * D + c];
}

__global__ void build_l11pad(const int* __restrict__ i11, const float* __restrict__ e11,
                             const float* __restrict__ pe, float* __restrict__ x) {
  const int bt = blockIdx.x, c = threadIdx.x;
  const float v = (c < 64) ? e11[(size_t)i11[bt] * 64 + c] : 0.0f;
  x[(size_t)bt * D + c] = v + pe[(bt % T) * D + c];
}

__global__ void build_cat(const float* __restrict__ hid, const int* __restrict__ idx,
                          const float* __restrict__ emb, int embdim,
                          float* __restrict__ feat, int fdim) {
  const int bt = blockIdx.x, c = threadIdx.x;          // fdim threads (320/384)
  const float v = (c < D) ? hid[(size_t)bt * D + c]
                          : emb[(size_t)idx[bt] * embdim + (c - D)];
  feat[(size_t)bt * fdim + c] = v;
}

__global__ void zero_kernel(float* __restrict__ p, size_t n) {
  for (size_t i = (size_t)blockIdx.x * blockDim.x + threadIdx.x; i < n;
       i += (size_t)gridDim.x * blockDim.x)
    p[i] = 0.0f;
}

__global__ void compute_last(const unsigned char* __restrict__ pad, int* __restrict__ last) {
  const int b = threadIdx.x;                           // 256 threads, 1 block
  int cnt = 0;
  for (int t = 0; t < T; ++t) cnt += pad[b * T + t] ? 0 : 1;
  last[b] = cnt - 1;
}

__global__ void gather_rows(const float* __restrict__ src, const int* __restrict__ last,
                            float* __restrict__ dst) {
  const int b = blockIdx.x, c = threadIdx.x;
  dst[(size_t)b * D + c] = src[((size_t)b * T + last[b]) * D + c];
}

// copy-attention: softmax over T scores, then scatter-add into copy_dist
__global__ __launch_bounds__(128)
void copy_attn_kernel(const float* __restrict__ Q, const float* __restrict__ K,
                      const int* __restrict__ xs, const unsigned char* __restrict__ pad,
                      float* __restrict__ cdist) {
  const int b = blockIdx.x, t = threadIdx.x;
  __shared__ float red[128];
  float s = -1e30f;
  if (t < T) {
    const float* q = Q + (size_t)b * D;
    const float* k = K + ((size_t)b * T + t) * D;
    float d = 0.0f;
    for (int c = 0; c < D; ++c) d += q[c] * k[c];
    s = d * (1.0f / 16.0f);                            // 1/sqrt(256)
    if (pad[b * T + t]) s = -1e9f;
  }
  red[t] = s; __syncthreads();
  for (int o = 64; o; o >>= 1) { if (t < o) red[t] = fmaxf(red[t], red[t + o]); __syncthreads(); }
  const float mx = red[0]; __syncthreads();
  const float e = (t < T) ? expf(s - mx) : 0.0f;
  red[t] = e; __syncthreads();
  for (int o = 64; o; o >>= 1) { if (t < o) red[t] += red[t + o]; __syncthreads(); }
  const float sum = red[0];
  if (t < T && !pad[b * T + t])
    atomicAdd(&cdist[(size_t)b * VX + xs[b * T + t]], e / sum);
}

__global__ void gate_kernel(const float* __restrict__ fs, const float* __restrict__ gw,
                            const float* __restrict__ gb, float* __restrict__ gate) {
  const int b = threadIdx.x;                           // 256 threads, 1 block
  float d = 0.0f;
  for (int c = 0; c < D; ++c) d += (gw[c] + gw[D + c]) * fs[(size_t)b * D + c];
  d += gb[0];
  gate[b] = 1.0f / (1.0f + expf(-d));
}

__global__ __launch_bounds__(256)
void row_stats_kernel(const float* __restrict__ gen, float* __restrict__ gmx,
                      float* __restrict__ gsum) {
  const int b = blockIdx.x, t = threadIdx.x;
  __shared__ float red[256];
  const float* r = gen + (size_t)b * VX;
  float m = -1e30f;
  for (int i = t; i < VX; i += 256) m = fmaxf(m, r[i]);
  red[t] = m; __syncthreads();
  for (int o = 128; o; o >>= 1) { if (t < o) red[t] = fmaxf(red[t], red[t + o]); __syncthreads(); }
  const float mx = red[0]; __syncthreads();
  float s = 0.0f;
  for (int i = t; i < VX; i += 256) s += expf(r[i] - mx);
  red[t] = s; __syncthreads();
  for (int o = 128; o; o >>= 1) { if (t < o) red[t] += red[t + o]; __syncthreads(); }
  if (t == 0) { gmx[b] = mx; gsum[b] = red[0]; }
}

__global__ void combine_kernel(const float* __restrict__ cdist, const float* __restrict__ gen,
                               const float* __restrict__ gmx, const float* __restrict__ gsum,
                               const float* __restrict__ gate, float* __restrict__ out) {
  const int b = blockIdx.y;
  const int i = blockIdx.x * 256 + threadIdx.x;
  if (i >= VX) return;
  const size_t o = (size_t)b * VX + i;
  const float g = gate[b];
  const float gp = expf(gen[o] - gmx[b]) / gsum[b];
  out[o] = logf(g * (cdist[o] + 1e-10f) + (1.0f - g) * gp + 1e-10f);
}

// ---------------- host-side helpers ----------------
struct LayerP {
  const __bf16 *in_w, *out_w, *w1, *w2;                 // bf16 weights
  const float *in_b, *out_b, *ln1_g, *ln1_b, *b1, *b2, *ln2_g, *ln2_b;
};

static inline void cvt(const float* a, __bf16* o, size_t n, hipStream_t s) {
  size_t blocks = (n + 255) / 256;
  if (blocks > 4096) blocks = 4096;
  cvt_bf16_kernel<<<(int)blocks, 256, 0, s>>>(a, o, n);
}

static inline void gemm(const __bf16* A, const __bf16* W, const float* bias,
                        const float* res, float* C, int M, int N, int K, int gelu,
                        hipStream_t s) {
  dim3 g((N + 63) / 64, M / 64);
  if (res)
    gemm_bf16_wmma<true, false><<<g, 128, 0, s>>>(A, W, bias, res, C, M, N, K);
  else if (gelu)
    gemm_bf16_wmma<false, true><<<g, 128, 0, s>>>(A, W, bias, nullptr, C, M, N, K);
  else
    gemm_bf16_wmma<false, false><<<g, 128, 0, s>>>(A, W, bias, nullptr, C, M, N, K);
}

// convert f32 A to bf16 scratch, then GEMM
static inline void gemm_f32A(const float* A, __bf16* ABF, const __bf16* W,
                             const float* bias, const float* res, float* C,
                             int M, int N, int K, int gelu, hipStream_t s) {
  cvt(A, ABF, (size_t)M * K, s);
  gemm(ABF, W, bias, res, C, M, N, K, gelu, s);
}

static void run_layer(const LayerP& L, float* x, float* qkv, float* ao, float* tmp,
                      __bf16* ABF, const unsigned char* pad, hipStream_t s) {
  gemm_f32A(x, ABF, L.in_w, L.in_b, nullptr, qkv, BT, 3 * D, D, 0, s);
  attn_kernel<<<dim3(B * H), 128, 0, s>>>(qkv, pad, ao);
  gemm_f32A(ao, ABF, L.out_w, L.out_b, x, tmp, BT, D, D, 0, s);
  ln_kernel<<<BT / 8, 256, 0, s>>>(tmp, L.ln1_g, L.ln1_b, x);
  gemm_f32A(x, ABF, L.w1, L.b1, nullptr, qkv, BT, 2 * D, D, 1, s);   // gelu
  gemm_f32A(qkv, ABF, L.w2, L.b2, x, tmp, BT, D, 2 * D, 0, s);
  ln_kernel<<<BT / 8, 256, 0, s>>>(tmp, L.ln2_g, L.ln2_b, x);
}

static void run_encoder(const LayerP* Ls, float* x, float* qkv, float* ao, float* tmp,
                        __bf16* ABF, const unsigned char* pad, hipStream_t s) {
  run_layer(Ls[0], x, qkv, ao, tmp, ABF, pad, s);
  run_layer(Ls[1], x, qkv, ao, tmp, ABF, pad, s);
}

extern "C" void kernel_launch(void* const* d_in, const int* in_sizes, int n_in,
                              void* d_out, int out_size, void* d_ws, size_t ws_size,
                              hipStream_t stream) {
  (void)in_sizes; (void)n_in; (void)out_size; (void)ws_size;

  // ---- inputs (setup_inputs() insertion order, params flattened in order) ----
  const int* l11_seq = (const int*)d_in[0];
  const int* l13_seq = (const int*)d_in[1];
  const int* l14_seq = (const int*)d_in[2];
  const int* X_seq   = (const int*)d_in[3];
  const int* user_sq = (const int*)d_in[4];
  const unsigned char* pad = (const unsigned char*)d_in[5];
  int p = 6;
  const float* user_emb = (const float*)d_in[p++];
  const float* emb_l11  = (const float*)d_in[p++];
  const float* emb_l13  = (const float*)d_in[p++];
  const float* emb_l14  = (const float*)d_in[p++];
  const float* emb_X    = (const float*)d_in[p++];
  const float* loc_w    = (const float*)d_in[p++];
  const float* loc_b    = (const float*)d_in[p++];
  const float* lw[2][4]; const float* lb[2][8];
  for (int l = 0; l < 2; ++l) {
    lw[l][0] = (const float*)d_in[p++];  lb[l][0] = (const float*)d_in[p++];  // in_w, in_b
    lw[l][1] = (const float*)d_in[p++];  lb[l][1] = (const float*)d_in[p++];  // out_w, out_b
    lb[l][2] = (const float*)d_in[p++];  lb[l][3] = (const float*)d_in[p++];  // ln1_g, ln1_b
    lw[l][2] = (const float*)d_in[p++];  lb[l][4] = (const float*)d_in[p++];  // w1, b1
    lw[l][3] = (const float*)d_in[p++];  lb[l][5] = (const float*)d_in[p++];  // w2, b2
    lb[l][6] = (const float*)d_in[p++];  lb[l][7] = (const float*)d_in[p++];  // ln2_g, ln2_b
  }
  const float* q_w  = (const float*)d_in[p++];  const float* q_b  = (const float*)d_in[p++];
  const float* k_w  = (const float*)d_in[p++];  const float* k_b  = (const float*)d_in[p++];
  const float* gw   = (const float*)d_in[p++];  const float* gb   = (const float*)d_in[p++];
  const float* p13_w = (const float*)d_in[p++]; const float* p13_b = (const float*)d_in[p++];
  const float* p14_w = (const float*)d_in[p++]; const float* p14_b = (const float*)d_in[p++];
  const float* pre_w = (const float*)d_in[p++]; const float* pre_b = (const float*)d_in[p++];
  const float* c11_w = (const float*)d_in[p++]; const float* c11_b = (const float*)d_in[p++];
  const float* c13_w = (const float*)d_in[p++]; const float* c13_b = (const float*)d_in[p++];
  const float* c14_w = (const float*)d_in[p++]; const float* c14_b = (const float*)d_in[p++];
  const float* cX_w  = (const float*)d_in[p++]; const float* cX_b  = (const float*)d_in[p++];

  // ---- outputs (flat concat in reference return order) ----
  float* out = (float*)d_out;
  float* out_l11  = out;
  float* out_l13  = out_l11 + (size_t)B * V11;
  float* out_l14  = out_l13 + (size_t)B * V13;
  float* out_X    = out_l14 + (size_t)B * V14;
  float* out_l11L = out_X   + (size_t)B * VX;
  float* out_l13L = out_l11L + (size_t)B * D;
  float* out_l14L = out_l13L + (size_t)B * D;
  float* out_fs   = out_l14L + (size_t)B * D;

  // ---- workspace layout: f32 region, then bf16 region ----
  float* w = (float*)d_ws;
  size_t o = 0;
  auto alloc = [&](size_t n) { float* r = w + o; o += (n + 63) & ~(size_t)63; return r; };
  float* PE   = alloc((size_t)T * D);
  float* FEAT = alloc((size_t)BT * 384);
  float* X1   = alloc((size_t)BT * D);        // activations
  float* QKV  = alloc((size_t)BT * 3 * D);    // qkv / ff1 / copy_dist (reused)
  float* AO   = alloc((size_t)BT * D);        // attn out / K-projection
  float* TMP  = alloc((size_t)BT * D);
  float* HIER = alloc((size_t)BT * D);        // l11_hidden / l13_hidden
  float* GEN  = alloc((size_t)B * VX);        // gen_logits
  float* QB   = alloc((size_t)B * D);
  float* HIDB = alloc((size_t)B * HID);
  float* GMX  = alloc(B);
  float* GSUM = alloc(B);
  float* GATE = alloc(B);
  int*   LAST = (int*)alloc(B);
  float* CDIST = QKV;                         // 10.24M floats fits in QKV region

  __bf16* bw = (__bf16*)(w + o);
  size_t ob = 0;
  auto balloc = [&](size_t n) { __bf16* r = bw + ob; ob += (n + 63) & ~(size_t)63; return r; };
  __bf16* ABF = balloc((size_t)BT * 512);     // activation bf16 scratch (max M*K)
  // bf16 weights, converted once per launch
  __bf16* loc_wb = balloc((size_t)D * 384);
  __bf16* q_wb   = balloc((size_t)D * D);
  __bf16* k_wb   = balloc((size_t)D * D);
  __bf16* p13_wb = balloc((size_t)D * 320);
  __bf16* p14_wb = balloc((size_t)D * 384);
  __bf16* pre_wb = balloc((size_t)HID * D);
  __bf16* c11_wb = balloc((size_t)V11 * HID);
  __bf16* c13_wb = balloc((size_t)V13 * HID);
  __bf16* c14_wb = balloc((size_t)V14 * HID);
  __bf16* cX_wb  = balloc((size_t)VX  * HID);
  LayerP Ls[2];
  for (int l = 0; l < 2; ++l) {
    Ls[l].in_w  = balloc((size_t)3 * D * D);
    Ls[l].out_w = balloc((size_t)D * D);
    Ls[l].w1    = balloc((size_t)2 * D * D);
    Ls[l].w2    = balloc((size_t)D * 2 * D);
    Ls[l].in_b = lb[l][0]; Ls[l].out_b = lb[l][1];
    Ls[l].ln1_g = lb[l][2]; Ls[l].ln1_b = lb[l][3];
    Ls[l].b1 = lb[l][4]; Ls[l].b2 = lb[l][5];
    Ls[l].ln2_g = lb[l][6]; Ls[l].ln2_b = lb[l][7];
  }

  // ---- 0. constants + one-time weight down-conversion ----
  pe_kernel<<<T, D, 0, stream>>>(PE);
  compute_last<<<1, B, 0, stream>>>(pad, LAST);
  cvt(loc_w, loc_wb, (size_t)D * 384, stream);
  cvt(q_w, q_wb, (size_t)D * D, stream);
  cvt(k_w, k_wb, (size_t)D * D, stream);
  cvt(p13_w, p13_wb, (size_t)D * 320, stream);
  cvt(p14_w, p14_wb, (size_t)D * 384, stream);
  cvt(pre_w, pre_wb, (size_t)HID * D, stream);
  cvt(c11_w, c11_wb, (size_t)V11 * HID, stream);
  cvt(c13_w, c13_wb, (size_t)V13 * HID, stream);
  cvt(c14_w, c14_wb, (size_t)V14 * HID, stream);
  cvt(cX_w, cX_wb, (size_t)VX * HID, stream);
  for (int l = 0; l < 2; ++l) {
    cvt(lw[l][0], (__bf16*)Ls[l].in_w,  (size_t)3 * D * D, stream);
    cvt(lw[l][1], (__bf16*)Ls[l].out_w, (size_t)D * D, stream);
    cvt(lw[l][2], (__bf16*)Ls[l].w1,    (size_t)2 * D * D, stream);
    cvt(lw[l][3], (__bf16*)Ls[l].w2,    (size_t)D * 2 * D, stream);
  }

  // ---- 1. main pass: embeddings -> loc proj -> +PE +user -> encoder ----
  build_feat<<<BT, 384, 0, stream>>>(l11_seq, l13_seq, l14_seq, X_seq,
                                     emb_l11, emb_l13, emb_l14, emb_X, FEAT);
  gemm_f32A(FEAT, ABF, loc_wb, loc_b, nullptr, X1, BT, D, 384, 0, stream);
  add_pe_user<<<BT, D, 0, stream>>>(X1, PE, user_emb, user_sq);
  run_encoder(Ls, X1, QKV, AO, TMP, ABF, pad, stream);

  // ---- 2. copy attention + X head (complete before QKV is reused) ----
  gather_rows<<<B, D, 0, stream>>>(X1, LAST, out_fs);               // final_state
  gemm_f32A(out_fs, ABF, q_wb, q_b, nullptr, QB, B, D, D, 0, stream);
  gemm_f32A(X1, ABF, k_wb, k_b, nullptr, AO, BT, D, D, 0, stream);  // K
  zero_kernel<<<2048, 256, 0, stream>>>(CDIST, (size_t)B * VX);
  copy_attn_kernel<<<B, 128, 0, stream>>>(QB, AO, X_seq, pad, CDIST);
  gemm_f32A(out_fs, ABF, pre_wb, pre_b, nullptr, HIDB, B, HID, D, 1, stream);
  gemm_f32A(HIDB, ABF, cX_wb, cX_b, nullptr, GEN, B, VX, HID, 0, stream);
  row_stats_kernel<<<B, 256, 0, stream>>>(GEN, GMX, GSUM);
  gate_kernel<<<1, B, 0, stream>>>(out_fs, gw, gb, GATE);
  combine_kernel<<<dim3((VX + 255) / 256, B), 256, 0, stream>>>(
      CDIST, GEN, GMX, GSUM, GATE, out_X);

  // ---- 3. hierarchical pass l11 ----
  build_l11pad<<<BT, D, 0, stream>>>(l11_seq, emb_l11, PE, X1);
  run_encoder(Ls, X1, QKV, AO, TMP, ABF, pad, stream);
  hipMemcpyAsync(HIER, X1, (size_t)BT * D * sizeof(float),
                 hipMemcpyDeviceToDevice, stream);
  gather_rows<<<B, D, 0, stream>>>(X1, LAST, out_l11L);
  gemm_f32A(out_l11L, ABF, pre_wb, pre_b, nullptr, HIDB, B, HID, D, 1, stream);
  gemm_f32A(HIDB, ABF, c11_wb, c11_b, nullptr, out_l11, B, V11, HID, 0, stream);

  // ---- 4. hierarchical pass l13 ----
  build_cat<<<BT, 320, 0, stream>>>(HIER, l13_seq, emb_l13, 64, FEAT, 320);
  gemm_f32A(FEAT, ABF, p13_wb, p13_b, nullptr, X1, BT, D, 320, 0, stream);
  add_pe<<<BT, D, 0, stream>>>(X1, PE);
  run_encoder(Ls, X1, QKV, AO, TMP, ABF, pad, stream);
  hipMemcpyAsync(HIER, X1, (size_t)BT * D * sizeof(float),
                 hipMemcpyDeviceToDevice, stream);
  gather_rows<<<B, D, 0, stream>>>(X1, LAST, out_l13L);
  gemm_f32A(out_l13L, ABF, pre_wb, pre_b, nullptr, HIDB, B, HID, D, 1, stream);
  gemm_f32A(HIDB, ABF, c13_wb, c13_b, nullptr, out_l13, B, V13, HID, 0, stream);

  // ---- 5. hierarchical pass l14 ----
  build_cat<<<BT, 384, 0, stream>>>(HIER, l14_seq, emb_l14, 128, FEAT, 384);
  gemm_f32A(FEAT, ABF, p14_wb, p14_b, nullptr, X1, BT, D, 384, 0, stream);
  add_pe<<<BT, D, 0, stream>>>(X1, PE);
  run_encoder(Ls, X1, QKV, AO, TMP, ABF, pad, stream);
  gather_rows<<<B, D, 0, stream>>>(X1, LAST, out_l14L);
  gemm_f32A(out_l14L, ABF, pre_wb, pre_b, nullptr, HIDB, B, HID, D, 1, stream);
  gemm_f32A(HIDB, ABF, c14_wb, c14_b, nullptr, out_l14, B, V14, HID, 0, stream);
}
